// MultiHeadAttention_59107339927887
// MI455X (gfx1250) — matrix operand order repaired and checked by reading, not compiled, not run on hardware
//
#include <hip/hip_runtime.h>
#include <hip/hip_bf16.h>

// ---------------------------------------------------------------------------
// MHA forward for MI455X (gfx1250, wave32, WMMA).
//   D_MODEL=1024, H=16, d_k=64, B=2, S=2048.
// All GEMMs run through v_wmma_f32_16x16x32_bf16 (fp32 accumulate).
// ---------------------------------------------------------------------------

#define DMODEL 1024
#define NHEADS 16
#define DKH    64
#define SEQ    2048
#define BATCH  2
#define MROWS  (BATCH * SEQ)                 // 4096 token rows
#define ACT_ELEMS ((size_t)MROWS * DMODEL)   // 4,194,304
#define W_ELEMS   ((size_t)DMODEL * DMODEL)  // 1,048,576

// log2(e) / sqrt(d_k): folded into Q so softmax uses raw v_exp_f32
#define QSCALE 0.18033688011112042f

typedef __attribute__((ext_vector_type(16))) __bf16        bf16x16;
typedef __attribute__((ext_vector_type(8)))  float         f32x8;
typedef __attribute__((ext_vector_type(4)))  unsigned int  u32x4;

union FragU {
    bf16x16      v;
    u32x4        q[2];
    unsigned int u[8];
};

__device__ __forceinline__ f32x8 wmma_bf16(bf16x16 a, bf16x16 b, f32x8 c) {
    // (neg_a, A, neg_b, B, c_mod, C, reuse_a, reuse_b)
    return __builtin_amdgcn_wmma_f32_16x16x32_bf16(false, a, false, b,
                                                   (short)0, c, false, false);
}

__device__ __forceinline__ f32x8 zero8() {
    f32x8 z = {0.f, 0.f, 0.f, 0.f, 0.f, 0.f, 0.f, 0.f};
    return z;
}

__device__ __forceinline__ unsigned int pack_bf16(float a, float b) {
    union { __bf16 h[2]; unsigned int u; } x;
    x.h[0] = (__bf16)a;
    x.h[1] = (__bf16)b;
    return x.u;
}

// Raw v_exp_f32: args are <= 0 here; sub-2^-126 probabilities may flush to 0,
// which is numerically irrelevant for softmax.  Avoids libm's range fixup.
__device__ __forceinline__ float fast_exp2(float x) {
    return __builtin_amdgcn_exp2f(x);
}

// A operand (M=16, K=32) from a row-major [rows x ldk] bf16 matrix.
// contiguous K order used here (same as B): lanes 0-15 -> M=lane, K=k0..k0+15;
//                  lanes 16-31 -> M=lane-16, K=k0+16..k0+31.
__device__ __forceinline__ bf16x16
load_a_frag(const __bf16* __restrict__ base, int ldk, int row0, int k0, int lane) {
    const int hh = lane >> 4;
    const int m  = lane & 15;
    const __bf16* p = base + (size_t)(row0 + m) * ldk + k0 + 16 * hh;
    FragU f;
    f.q[0] = *(const u32x4*)(p);        // 8 bf16
    f.q[1] = *(const u32x4*)(p + 8);    // 8 bf16, K+8
    return f.v;
}

// B operand (K=32, N=16) in NT form: rows of `base` are the 16 columns of B.
// contiguous K order used here (same as A): lanes 0-15 -> col=lane,   K=k0..k0+15;
//                      lanes 16-31 -> col=lane-16, K=k0+16..k0+31.
__device__ __forceinline__ bf16x16
load_b_frag(const __bf16* __restrict__ base, int ldk, int col0, int k0, int lane) {
    const int hh = lane >> 4;
    const int n  = lane & 15;
    const __bf16* p = base + (size_t)(col0 + n) * ldk + k0 + 16 * hh;
    FragU f;
    f.q[0] = *(const u32x4*)(p);        // 16 contiguous bf16
    f.q[1] = *(const u32x4*)(p + 8);
    return f.v;
}

// ---------------------------------------------------------------------------
// fp32 -> bf16 cast, 4 elements / thread (b128 in, b64 out)
// ---------------------------------------------------------------------------
__global__ void cast_f32_bf16_x4(const float* __restrict__ in,
                                 __bf16* __restrict__ out, int n4) {
    int i = blockIdx.x * blockDim.x + threadIdx.x;
    if (i < n4) {
        float4 f = ((const float4*)in)[i];
        union { __bf16 h[4]; unsigned long long u; } o;
        o.h[0] = (__bf16)f.x; o.h[1] = (__bf16)f.y;
        o.h[2] = (__bf16)f.z; o.h[3] = (__bf16)f.w;
        ((unsigned long long*)out)[i] = o.u;
    }
}

// ---------------------------------------------------------------------------
// NT GEMM: C[M=4096, N=1024] = A(bf16, MxK) * W(bf16, NxK)^T + bias
// Each wave computes a 32x64 tile: 2 A-frags x 4 B-frags -> 8 wmma / K-step
// (every fragment load feeds >= 2 WMMAs).
// MODE 0: Q-proj -> Qh (b,h,s,dk) bf16, scaled by log2e/sqrt(dk)
// MODE 1: K-proj -> Kh (b,h,s,dk) bf16
// MODE 2: V-proj -> Vt (b,h,dk,s) bf16  (transposed so P*V is NT)
// MODE 3: O-proj -> out (b,s,d) fp32
// ---------------------------------------------------------------------------
template <int MODE>
__global__ __launch_bounds__(256) void gemm_nt_kernel(
    const __bf16* __restrict__ A, const __bf16* __restrict__ W,
    const float* __restrict__ bias, void* __restrict__ Cout) {
    const int lane = threadIdx.x & 31;
    const int wave = blockIdx.x * (blockDim.x >> 5) + (threadIdx.x >> 5);
    const int m0 = (wave & 127) << 5;   // 128 M-tiles of 32
    const int n0 = (wave >> 7) << 6;    // 16 N-panels of 64

    f32x8 acc[2][4] = {{zero8(), zero8(), zero8(), zero8()},
                       {zero8(), zero8(), zero8(), zero8()}};

    for (int k0 = 0; k0 < DMODEL; k0 += 32) {
        bf16x16 af0 = load_a_frag(A, DMODEL, m0,      k0, lane);
        bf16x16 af1 = load_a_frag(A, DMODEL, m0 + 16, k0, lane);
#pragma unroll
        for (int t = 0; t < 4; ++t) {
            bf16x16 bfr = load_b_frag(W, DMODEL, n0 + 16 * t, k0, lane);
            acc[0][t] = wmma_bf16(af0, bfr, acc[0][t]);
            acc[1][t] = wmma_bf16(af1, bfr, acc[1][t]);
        }
    }

    // C/D layout: VGPR r: lanes 0-15 -> M=r, N=lane; lanes 16-31 -> M=r+8.
    const int hh = lane >> 4;
    const int nn = lane & 15;
#pragma unroll
    for (int i = 0; i < 2; ++i) {
#pragma unroll
        for (int t = 0; t < 4; ++t) {
#pragma unroll
            for (int r = 0; r < 8; ++r) {
                const int m = m0 + 16 * i + r + 8 * hh;
                const int n = n0 + 16 * t + nn;
                float vv = acc[i][t][r] + bias[n];
                if (MODE == 3) {
                    ((float*)Cout)[(size_t)m * DMODEL + n] = vv;
                } else {
                    const int b = m >> 11, s = m & (SEQ - 1);
                    const int h = n >> 6,  d = n & (DKH - 1);
                    __bf16* o = (__bf16*)Cout;
                    if (MODE == 0)
                        o[((size_t)(b * NHEADS + h) * SEQ + s) * DKH + d] =
                            (__bf16)(vv * QSCALE);
                    else if (MODE == 1)
                        o[((size_t)(b * NHEADS + h) * SEQ + s) * DKH + d] =
                            (__bf16)vv;
                    else
                        o[((size_t)(b * NHEADS + h) * DKH + d) * SEQ + s] =
                            (__bf16)vv;
                }
            }
        }
    }
}

// ---------------------------------------------------------------------------
// Flash attention, transposed-score formulation.
// One wave per (b, h, 16-query tile).  Inner loop: 32 keys / iteration.
//   S^T tile   = K_tile(16k x 64) * Q^T(64 x 16q)         -> 4 wmma
//   ctx^T tile = V^T(64d x 32k)   * P^T(32k x 16q)        -> 4 wmma
// Softmax rows (fixed query) live in one lane + its xor-16 partner, so the
// online max/sum needs a single __shfl_xor(16); the ctx rescale by alpha is a
// per-lane scalar.  P^T D-layout -> B-layout needs 8 packed xor-16 shuffles.
// ---------------------------------------------------------------------------
__global__ __launch_bounds__(256) void flash_attn_kernel(
    const __bf16* __restrict__ Qh, const __bf16* __restrict__ Kh,
    const __bf16* __restrict__ Vt, __bf16* __restrict__ Ctx2) {
    const int lane = threadIdx.x & 31;
    const int wave = blockIdx.x * (blockDim.x >> 5) + (threadIdx.x >> 5);
    const int qt = wave & 127;          // 128 query tiles of 16
    const int bh = wave >> 7;           // b*16 + h
    const int m0 = qt << 4;
    const int hh = lane >> 4;
    const int nn = lane & 15;

    const __bf16* Qp = Qh + ((size_t)bh * SEQ + m0) * DKH;  // 16 x 64
    const __bf16* Kp = Kh + (size_t)bh * SEQ * DKH;         // S  x 64
    const __bf16* Vp = Vt + (size_t)bh * DKH * SEQ;         // 64 x S

    // Q^T as B operand (col n = query row m0+n), preloaded for dk 0..31,32..63
    const bf16x16 bq0 = load_b_frag(Qp, DKH, 0, 0, lane);
    const bf16x16 bq1 = load_b_frag(Qp, DKH, 0, 32, lane);

    f32x8 ctx[4] = {zero8(), zero8(), zero8(), zero8()};
    float mrow = -3.0e38f;   // running max (base-2 scaled domain)
    float lrow = 0.0f;       // running denominator

    for (int kb = 0; kb < SEQ; kb += 32) {
        // ---- S^T tiles: keys kb..kb+15 (s0) and kb+16..kb+31 (s1) ----
        f32x8 s0 = zero8(), s1 = zero8();
        {
            bf16x16 a = load_a_frag(Kp, DKH, kb, 0, lane);
            s0 = wmma_bf16(a, bq0, s0);
            a = load_a_frag(Kp, DKH, kb, 32, lane);
            s0 = wmma_bf16(a, bq1, s0);
            a = load_a_frag(Kp, DKH, kb + 16, 0, lane);
            s1 = wmma_bf16(a, bq0, s1);
            a = load_a_frag(Kp, DKH, kb + 16, 32, lane);
            s1 = wmma_bf16(a, bq1, s1);
        }

        // ---- online softmax stats (per-lane query = nn) ----
        float tm = -3.0e38f;
#pragma unroll
        for (int r = 0; r < 8; ++r) tm = fmaxf(tm, fmaxf(s0[r], s1[r]));
        tm = fmaxf(tm, __shfl_xor(tm, 16, 32));
        const float mnew  = fmaxf(mrow, tm);
        const float alpha = fast_exp2(mrow - mnew);
        mrow = mnew;

        float p0[8], p1[8], psum = 0.f;
#pragma unroll
        for (int r = 0; r < 8; ++r) {
            p0[r] = fast_exp2(s0[r] - mnew);
            p1[r] = fast_exp2(s1[r] - mnew);
            psum += p0[r] + p1[r];
        }
        psum += __shfl_xor(psum, 16, 32);
        lrow = lrow * alpha + psum;

        // ---- P^T register transpose: D-layout -> B-layout (keys = wmma K) ----
        unsigned int pk0[4], pk1[4];
#pragma unroll
        for (int r = 0; r < 4; ++r) {
            pk0[r] = pack_bf16(p0[2 * r], p0[2 * r + 1]);
            pk1[r] = pack_bf16(p1[2 * r], p1[2 * r + 1]);
        }
        FragU bp;
#pragma unroll
        for (int r = 0; r < 4; ++r) {
            const unsigned int spk0 = __shfl_xor(pk0[r], 16, 32);
            const unsigned int spk1 = __shfl_xor(pk1[r], 16, 32);
            bp.u[r]     = (hh == 0) ? pk0[r] : spk1;   // wmma-K 0..7 of this half
            bp.u[4 + r] = (hh == 0) ? spk0 : pk1[r];   // wmma-K 8..15
        }

        // ---- ctx^T += V^T * P^T, with per-lane scalar rescale by alpha ----
#pragma unroll
        for (int t = 0; t < 4; ++t) {
            ctx[t] = ctx[t] * alpha;                   // v_pk_mul_f32
            bf16x16 av = load_a_frag(Vp, SEQ, 16 * t, kb, lane);
            ctx[t] = wmma_bf16(av, bp.v, ctx[t]);
        }
    }

    // ---- epilogue: ctx^T[d][q] / lrow -> Ctx2[(b,s),(h*64+d)] as bf16 ----
    const float rinv = 1.0f / lrow;
    const int b = bh >> 4, h = bh & 15;
    __bf16* crow =
        Ctx2 + ((size_t)b * SEQ + m0 + nn) * DMODEL + h * DKH + 8 * hh;
#pragma unroll
    for (int t = 0; t < 4; ++t) {
        FragU o;
#pragma unroll
        for (int r = 0; r < 4; ++r)
            o.u[r] = pack_bf16(ctx[t][2 * r] * rinv, ctx[t][2 * r + 1] * rinv);
        *(u32x4*)(crow + 16 * t) = o.q[0];   // 8 contiguous bf16 (b128 store)
    }
}

// ---------------------------------------------------------------------------
extern "C" void kernel_launch(void* const* d_in, const int* in_sizes, int n_in,
                              void* d_out, int out_size, void* d_ws,
                              size_t ws_size, hipStream_t stream) {
    (void)in_sizes; (void)n_in; (void)out_size; (void)ws_size;
    const float* q    = (const float*)d_in[0];
    const float* k    = (const float*)d_in[1];
    const float* v    = (const float*)d_in[2];
    const float* Wq_w = (const float*)d_in[3];
    const float* Wq_b = (const float*)d_in[4];
    const float* Wk_w = (const float*)d_in[5];
    const float* Wk_b = (const float*)d_in[6];
    const float* Wv_w = (const float*)d_in[7];
    const float* Wv_b = (const float*)d_in[8];
    const float* Wo_w = (const float*)d_in[9];
    const float* Wo_b = (const float*)d_in[10];

    // Workspace layout (bf16): 3 activations + 4 weights + Qh,Kh,Vt,Ctx = 64 MiB
    __bf16* Xq  = (__bf16*)d_ws;
    __bf16* Xk  = Xq + ACT_ELEMS;
    __bf16* Xv  = Xk + ACT_ELEMS;
    __bf16* Wqb = Xv + ACT_ELEMS;
    __bf16* Wkb = Wqb + W_ELEMS;
    __bf16* Wvb = Wkb + W_ELEMS;
    __bf16* Wob = Wvb + W_ELEMS;
    __bf16* Qhp = Wob + W_ELEMS;
    __bf16* Khp = Qhp + ACT_ELEMS;
    __bf16* Vtp = Khp + ACT_ELEMS;
    __bf16* Ctx = Vtp + ACT_ELEMS;

    const int actB = (int)(ACT_ELEMS / 4 / 256);  // 4096 blocks
    const int wB   = (int)(W_ELEMS / 4 / 256);    // 1024 blocks
    cast_f32_bf16_x4<<<actB, 256, 0, stream>>>(q, Xq, (int)(ACT_ELEMS / 4));
    cast_f32_bf16_x4<<<actB, 256, 0, stream>>>(k, Xk, (int)(ACT_ELEMS / 4));
    cast_f32_bf16_x4<<<actB, 256, 0, stream>>>(v, Xv, (int)(ACT_ELEMS / 4));
    cast_f32_bf16_x4<<<wB, 256, 0, stream>>>(Wq_w, Wqb, (int)(W_ELEMS / 4));
    cast_f32_bf16_x4<<<wB, 256, 0, stream>>>(Wk_w, Wkb, (int)(W_ELEMS / 4));
    cast_f32_bf16_x4<<<wB, 256, 0, stream>>>(Wv_w, Wvb, (int)(W_ELEMS / 4));
    cast_f32_bf16_x4<<<wB, 256, 0, stream>>>(Wo_w, Wob, (int)(W_ELEMS / 4));

    // GEMMs: 2048 waves (32x64 tiles) = 256 blocks; flash: 4096 waves = 512
    gemm_nt_kernel<0><<<256, 256, 0, stream>>>(Xq, Wqb, Wq_b, (void*)Qhp);
    gemm_nt_kernel<1><<<256, 256, 0, stream>>>(Xk, Wkb, Wk_b, (void*)Khp);
    gemm_nt_kernel<2><<<256, 256, 0, stream>>>(Xv, Wvb, Wv_b, (void*)Vtp);
    flash_attn_kernel<<<512, 256, 0, stream>>>(Qhp, Khp, Vtp, Ctx);
    gemm_nt_kernel<3><<<256, 256, 0, stream>>>(Ctx, Wob, Wo_b, d_out);
}
